// SnapshotGNN_76836964926221
// MI455X (gfx1250) — compile-verified, matrix operand order checked by reading, not run on hardware
//
#include <hip/hip_runtime.h>
#include <hip/hip_bf16.h>
#include <math.h>

typedef __attribute__((ext_vector_type(16))) __bf16 v16bf;
typedef __attribute__((ext_vector_type(8)))  __bf16 v8bf;
typedef __attribute__((ext_vector_type(8)))  float  v8f;
typedef __attribute__((ext_vector_type(4)))  float  v4f;

// ---------------------------------------------------------------------------
// Weight conversion: f32 (in x out) -> bf16 transposed (out x in), plus
// concatenated [Wq|Wk|Wv|Ws] (512 x 128 per layer) and concatenated biases.
// ---------------------------------------------------------------------------
__global__ void convert_weights(
    const float* __restrict__ Wq, const float* __restrict__ Wk,
    const float* __restrict__ Wv, const float* __restrict__ Ws,
    const float* __restrict__ We, const float* __restrict__ Wo,
    const float* __restrict__ bq, const float* __restrict__ bk,
    const float* __restrict__ bv, const float* __restrict__ bs,
    __bf16* __restrict__ wt_qkvs, float* __restrict__ b_qkvs,
    __bf16* __restrict__ wt_e, __bf16* __restrict__ wt_o)
{
    int idx = blockIdx.x * blockDim.x + threadIdx.x;
    if (idx < 2 * 512 * 128) {               // [Wq|Wk|Wv|Ws] per layer, transposed
        int l = idx >> 16, r = idx & 65535;
        int n = r >> 7, k = r & 127;
        int part = n >> 7, col = n & 127;
        const float* W = (part == 0) ? Wq : (part == 1) ? Wk : (part == 2) ? Wv : Ws;
        wt_qkvs[idx] = (__bf16)W[l * 16384 + k * 128 + col];
    }
    if (idx < 2 * 128 * 128) {               // We per layer, transposed
        int l = idx >> 14, r = idx & 16383;
        int n = r >> 7, k = r & 127;
        wt_e[idx] = (__bf16)We[l * 16384 + k * 128 + n];
    }
    if (idx < 128 * 128) {                   // Wo, transposed
        int n = idx >> 7, k = idx & 127;
        wt_o[idx] = (__bf16)Wo[k * 128 + n];
    }
    if (idx < 2 * 512) {                     // concat bias [bq|bk|bv|bs]
        int l = idx >> 9, n = idx & 511;
        int part = n >> 7, col = n & 127;
        const float* b = (part == 0) ? bq : (part == 1) ? bk : (part == 2) ? bv : bs;
        b_qkvs[idx] = b[l * 128 + col];
    }
}

__global__ void fill_u32(unsigned* __restrict__ p, unsigned v, int n)
{
    int i = blockIdx.x * blockDim.x + threadIdx.x;
    int stride = gridDim.x * blockDim.x;
    for (; i < n; i += stride) p[i] = v;
}

// ---------------------------------------------------------------------------
// WMMA bf16 GEMM:  C[M x NOUT] = A[M x 128] * Wt^T  (+ bias)
// A is f32 row-major (converted to bf16 in-register), Wt is bf16 (NOUT x 128).
// 16-bit A-matrix 16x32 lane layout: lanes 0-15 -> K=0..7 & 16..23,
//                                    lanes 16-31 -> K=8..15 & 24..31.
// Fast path (full 16-row tile): unconditional coalesced stores, no exec ladder.
// ---------------------------------------------------------------------------
__device__ __forceinline__ v16bf load_a_f32(const float* __restrict__ p)
{
    v4f x0 = *(const v4f*)(p);
    v4f x1 = *(const v4f*)(p + 4);
    v4f x2 = *(const v4f*)(p + 16);
    v4f x3 = *(const v4f*)(p + 20);
    v16bf a;
#pragma unroll
    for (int i = 0; i < 4; i++) {
        a[i]      = (__bf16)x0[i];
        a[4 + i]  = (__bf16)x1[i];
        a[8 + i]  = (__bf16)x2[i];
        a[12 + i] = (__bf16)x3[i];
    }
    return a;
}

__device__ __forceinline__ v16bf load_b_bf16(const __bf16* __restrict__ p)
{
    v8bf lo = *(const v8bf*)(p);
    v8bf hi = *(const v8bf*)(p + 16);
    v16bf b;
#pragma unroll
    for (int i = 0; i < 8; i++) { b[i] = lo[i]; b[8 + i] = hi[i]; }
    return b;
}

template <int NOUT, bool BF16OUT>
__global__ __launch_bounds__(256)
void gemm_bf16_wmma(const float* __restrict__ A, int M,
                    const __bf16* __restrict__ Wt,
                    const float* __restrict__ bias,
                    float* __restrict__ Cf, __bf16* __restrict__ Cb)
{
    const int lane = threadIdx.x & 31;
    const int wave = threadIdx.x >> 5;
    const int r0 = (blockIdx.x * 8 + wave) * 16;
    if (r0 >= M) return;                       // wave-uniform: EXEC stays full
    const int half = lane >> 4;
    const int l15 = lane & 15;
    int rowA = r0 + l15;
    if (rowA > M - 1) rowA = M - 1;            // clamp (tail rows discarded at store)
    const float* arow = A + (size_t)rowA * 128 + half * 8;
    v16bf a[4];
#pragma unroll
    for (int kt = 0; kt < 4; kt++) a[kt] = load_a_f32(arow + kt * 32);

    const int mBase = r0 + half * 8;

    if (r0 + 16 <= M) {
        // ---- full tile: no per-lane guards ----
#pragma unroll 1
        for (int nt = 0; nt < NOUT / 16; nt++) {
            const int n = nt * 16 + l15;
            v8f acc;
            float bval = bias ? bias[n] : 0.f;
#pragma unroll
            for (int i = 0; i < 8; i++) acc[i] = bval;
            const __bf16* wrow = Wt + (size_t)n * 128 + half * 8;
            v16bf b[4];
#pragma unroll
            for (int kt = 0; kt < 4; kt++) b[kt] = load_b_bf16(wrow + kt * 32);
#pragma unroll
            for (int kt = 0; kt < 4; kt++)
                acc = __builtin_amdgcn_wmma_f32_16x16x32_bf16(
                    false, a[kt], false, b[kt], (short)0, acc, false, false);
#pragma unroll
            for (int v = 0; v < 8; v++) {
                int row = mBase + v;
                if (BF16OUT) Cb[(size_t)row * NOUT + n] = (__bf16)acc[v];
                else         Cf[(size_t)row * NOUT + n] = acc[v];
            }
        }
    } else {
        // ---- tail tile: guarded stores ----
#pragma unroll 1
        for (int nt = 0; nt < NOUT / 16; nt++) {
            const int n = nt * 16 + l15;
            v8f acc;
            float bval = bias ? bias[n] : 0.f;
#pragma unroll
            for (int i = 0; i < 8; i++) acc[i] = bval;
            const __bf16* wrow = Wt + (size_t)n * 128 + half * 8;
            v16bf b[4];
#pragma unroll
            for (int kt = 0; kt < 4; kt++) b[kt] = load_b_bf16(wrow + kt * 32);
#pragma unroll
            for (int kt = 0; kt < 4; kt++)
                acc = __builtin_amdgcn_wmma_f32_16x16x32_bf16(
                    false, a[kt], false, b[kt], (short)0, acc, false, false);
#pragma unroll
            for (int v = 0; v < 8; v++) {
                int row = mBase + v;
                if (row < M) {
                    if (BF16OUT) Cb[(size_t)row * NOUT + n] = (__bf16)acc[v];
                    else         Cf[(size_t)row * NOUT + n] = acc[v];
                }
            }
        }
    }
}

// ---------------------------------------------------------------------------
// Scatter-softmax attention, 3 passes over (edge, head). Ordered-uint float
// encoding for atomic max (init 0 encodes below every real float).
// ---------------------------------------------------------------------------
__global__ void edge_logits(const int* __restrict__ ei,
                            const float* __restrict__ QKVS,
                            const __bf16* __restrict__ Ebf,
                            float* __restrict__ P,
                            unsigned* __restrict__ Mx, int Ecnt)
{
    int idx = blockIdx.x * blockDim.x + threadIdx.x;
    if (idx >= Ecnt * 4) return;
    int edge = idx >> 2, h = idx & 3;
    int s = ei[edge], d = ei[Ecnt + edge];
    const float*  qp = QKVS + (size_t)d * 512 + h * 32;
    const float*  kp = QKVS + (size_t)s * 512 + 128 + h * 32;
    const __bf16* ep = Ebf + (size_t)edge * 128 + h * 32;
    float acc = 0.f;
#pragma unroll
    for (int c0 = 0; c0 < 32; c0 += 8) {
        v4f  q0 = *(const v4f*)(qp + c0);
        v4f  q1 = *(const v4f*)(qp + c0 + 4);
        v4f  k0 = *(const v4f*)(kp + c0);
        v4f  k1 = *(const v4f*)(kp + c0 + 4);
        v8bf e8 = *(const v8bf*)(ep + c0);
#pragma unroll
        for (int i = 0; i < 4; i++) acc += q0[i] * (k0[i] + (float)e8[i]);
#pragma unroll
        for (int i = 0; i < 4; i++) acc += q1[i] * (k1[i] + (float)e8[4 + i]);
    }
    float logit = acc * 0.17677669529663687f;   // 1/sqrt(32)
    P[idx] = logit;
    unsigned bits = __float_as_uint(logit);
    unsigned enc = (bits & 0x80000000u) ? ~bits : (bits | 0x80000000u);
    atomicMax(&Mx[d * 4 + h], enc);
}

__global__ void edge_exp(const int* __restrict__ ei,
                         float* __restrict__ P,
                         const unsigned* __restrict__ Mx,
                         float* __restrict__ Dn, int Ecnt)
{
    int idx = blockIdx.x * blockDim.x + threadIdx.x;
    if (idx >= Ecnt * 4) return;
    int edge = idx >> 2, h = idx & 3;
    int d = ei[Ecnt + edge];
    unsigned enc = Mx[d * 4 + h];
    float m = (enc & 0x80000000u) ? __uint_as_float(enc ^ 0x80000000u)
                                  : __uint_as_float(~enc);
    float p = __expf(P[idx] - m);
    P[idx] = p;
    __hip_atomic_fetch_add(&Dn[d * 4 + h], p, __ATOMIC_RELAXED, __HIP_MEMORY_SCOPE_AGENT);
}

__global__ void edge_scatter(const int* __restrict__ ei,
                             const float* __restrict__ P,
                             const float* __restrict__ Dn,
                             const float* __restrict__ QKVS,
                             const __bf16* __restrict__ Ebf,
                             float* __restrict__ Agg, int Ecnt)
{
    int idx = blockIdx.x * blockDim.x + threadIdx.x;
    if (idx >= Ecnt * 4) return;
    int edge = idx >> 2, h = idx & 3;
    int s = ei[edge], d = ei[Ecnt + edge];
    float alpha = P[idx] / (Dn[d * 4 + h] + 1e-16f);
    const float*  vp = QKVS + (size_t)s * 512 + 256 + h * 32;
    const __bf16* ep = Ebf + (size_t)edge * 128 + h * 32;
    float* op = Agg + (size_t)d * 128 + h * 32;
#pragma unroll
    for (int c = 0; c < 32; c += 4) {
        v4f vv = *(const v4f*)(vp + c);
#pragma unroll
        for (int i = 0; i < 4; i++) {
            float msg = alpha * (vv[i] + (float)ep[c + i]);
            __hip_atomic_fetch_add(op + c + i, msg, __ATOMIC_RELAXED,
                                   __HIP_MEMORY_SCOPE_AGENT);
        }
    }
}

// ---------------------------------------------------------------------------
// Per-node epilogue: beta gate, GELU (exact), LayerNorm, optional residual.
// One wave32 per node, 4 channels per lane, shuffle reductions.
// ---------------------------------------------------------------------------
__global__ void node_update(const float* __restrict__ Agg,
                            const float* __restrict__ QKVS,
                            const float* __restrict__ Wb,
                            const float* __restrict__ lng,
                            const float* __restrict__ lnb,
                            const float* __restrict__ addTo,
                            float* __restrict__ Hout, int Nn)
{
    int lane = threadIdx.x & 31;
    int node = blockIdx.x * 8 + (threadIdx.x >> 5);
    if (node >= Nn) return;
    int c = lane * 4;
    v4f o  = *(const v4f*)(Agg + (size_t)node * 128 + c);
    v4f xr = *(const v4f*)(QKVS + (size_t)node * 512 + 384 + c);
    v4f w0 = *(const v4f*)(Wb + c);
    v4f w1 = *(const v4f*)(Wb + 128 + c);
    v4f w2 = *(const v4f*)(Wb + 256 + c);
    float part = 0.f;
#pragma unroll
    for (int i = 0; i < 4; i++)
        part += o[i] * w0[i] + xr[i] * w1[i] + (o[i] - xr[i]) * w2[i];
#pragma unroll
    for (int m = 16; m >= 1; m >>= 1) part += __shfl_xor(part, m, 32);
    float beta = 1.f / (1.f + __expf(-part));
    v4f h;
#pragma unroll
    for (int i = 0; i < 4; i++) h[i] = beta * xr[i] + (1.f - beta) * o[i];
#pragma unroll
    for (int i = 0; i < 4; i++)
        h[i] = 0.5f * h[i] * (1.f + erff(h[i] * 0.70710678118654752f));
    float s = h[0] + h[1] + h[2] + h[3];
#pragma unroll
    for (int m = 16; m >= 1; m >>= 1) s += __shfl_xor(s, m, 32);
    float mu = s * (1.f / 128.f);
    float vs = 0.f;
#pragma unroll
    for (int i = 0; i < 4; i++) { float dd = h[i] - mu; vs += dd * dd; }
#pragma unroll
    for (int m = 16; m >= 1; m >>= 1) vs += __shfl_xor(vs, m, 32);
    float inv = rsqrtf(vs * (1.f / 128.f) + 1e-5f);
    v4f g4 = *(const v4f*)(lng + c);
    v4f b4 = *(const v4f*)(lnb + c);
    v4f y;
#pragma unroll
    for (int i = 0; i < 4; i++) y[i] = (h[i] - mu) * inv * g4[i] + b4[i];
    if (addTo) {
        v4f a = *(const v4f*)(addTo + (size_t)node * 128 + c);
#pragma unroll
        for (int i = 0; i < 4; i++) y[i] += a[i];
    }
    *(v4f*)(Hout + (size_t)node * 128 + c) = y;
}

__global__ void final_gate(const float* __restrict__ x,
                           const float* __restrict__ OUT,
                           const float* __restrict__ Wg,
                           const float* __restrict__ bg,
                           float* __restrict__ out, int Nn)
{
    int lane = threadIdx.x & 31;
    int node = blockIdx.x * 8 + (threadIdx.x >> 5);
    if (node >= Nn) return;
    int c = lane * 4;
    v4f xv = *(const v4f*)(x + (size_t)node * 128 + c);
    v4f wg = *(const v4f*)(Wg + c);
    float p = xv[0] * wg[0] + xv[1] * wg[1] + xv[2] * wg[2] + xv[3] * wg[3];
#pragma unroll
    for (int m = 16; m >= 1; m >>= 1) p += __shfl_xor(p, m, 32);
    float g = 1.f / (1.f + __expf(-(p + bg[0])));
    v4f ov = *(const v4f*)(OUT + (size_t)node * 128 + c);
    v4f r;
#pragma unroll
    for (int i = 0; i < 4; i++) r[i] = g * xv[i] + (1.f - g) * ov[i];
    *(v4f*)(out + (size_t)node * 128 + c) = r;
}

// ---------------------------------------------------------------------------
extern "C" void kernel_launch(void* const* d_in, const int* in_sizes, int n_in,
                              void* d_out, int out_size, void* d_ws, size_t ws_size,
                              hipStream_t stream)
{
    (void)in_sizes; (void)n_in; (void)out_size; (void)ws_size;
    const int N = 100000, E = 800000;
    const float* x   = (const float*)d_in[0];
    const int*   ei  = (const int*)  d_in[1];
    const float* ea  = (const float*)d_in[2];
    const float* Wq  = (const float*)d_in[3];
    const float* bq  = (const float*)d_in[4];
    const float* Wk  = (const float*)d_in[5];
    const float* bk  = (const float*)d_in[6];
    const float* Wv  = (const float*)d_in[7];
    const float* bv  = (const float*)d_in[8];
    const float* We  = (const float*)d_in[9];
    const float* Ws  = (const float*)d_in[10];
    const float* bs  = (const float*)d_in[11];
    const float* Wb  = (const float*)d_in[12];
    const float* lng = (const float*)d_in[13];
    const float* lnb = (const float*)d_in[14];
    const float* Wo  = (const float*)d_in[15];
    const float* bo  = (const float*)d_in[16];
    const float* Wg  = (const float*)d_in[17];
    const float* bg  = (const float*)d_in[18];
    float* out = (float*)d_out;

    char* ws = (char*)d_ws;
    size_t off = 0;
    auto take = [&](size_t bytes) -> char* {
        char* p = ws + off;
        off += (bytes + 255) & ~(size_t)255;
        return p;
    };
    __bf16* wt_qkvs = (__bf16*)take((size_t)2 * 512 * 128 * 2);
    float*  b_qkvs  = (float*) take((size_t)2 * 512 * 4);
    __bf16* wt_e    = (__bf16*)take((size_t)2 * 128 * 128 * 2);
    __bf16* wt_o    = (__bf16*)take((size_t)128 * 128 * 2);
    float*  QKVS    = (float*) take((size_t)N * 512 * 4);   // [Q|K|V|Xr] per node
    __bf16* Ebf     = (__bf16*)take((size_t)E * 128 * 2);   // edge features (bf16)
    float*  P       = (float*) take((size_t)E * 4 * 4);     // logits -> probs
    unsigned* Mx    = (unsigned*)take((size_t)N * 4 * 4);   // per (node,head) max (encoded)
    float*  Dn      = (float*) take((size_t)N * 4 * 4);     // softmax denom
    float*  Agg     = (float*) take((size_t)N * 128 * 4);   // aggregated messages
    float*  H1      = (float*) take((size_t)N * 128 * 4);   // layer-1 output
    float*  Hc      = (float*) take((size_t)N * 128 * 4);   // H1 + layer-2 output
    float*  OUTb    = (float*) take((size_t)N * 128 * 4);   // Hc @ Wo + bo

    dim3 blk(256);
    convert_weights<<<512, blk, 0, stream>>>(Wq, Wk, Wv, Ws, We, Wo, bq, bk, bv, bs,
                                             wt_qkvs, b_qkvs, wt_e, wt_o);

    const int gemmN = (N + 127) / 128;
    const int gemmE = (E + 127) / 128;
    const int nodeG = (N + 7) / 8;
    const int edgeG = (E * 4 + 255) / 256;

    for (int l = 0; l < 2; l++) {
        const float* Xin = (l == 0) ? x : H1;
        gemm_bf16_wmma<512, false><<<gemmN, blk, 0, stream>>>(
            Xin, N, wt_qkvs + (size_t)l * 512 * 128, b_qkvs + l * 512, QKVS, nullptr);
        gemm_bf16_wmma<128, true><<<gemmE, blk, 0, stream>>>(
            ea, E, wt_e + (size_t)l * 128 * 128, nullptr, nullptr, Ebf);
        fill_u32<<<1024, blk, 0, stream>>>(Mx, 0u, N * 4);
        fill_u32<<<1024, blk, 0, stream>>>((unsigned*)Dn, 0u, N * 4);
        fill_u32<<<2048, blk, 0, stream>>>((unsigned*)Agg, 0u, N * 128);
        edge_logits<<<edgeG, blk, 0, stream>>>(ei, QKVS, Ebf, P, Mx, E);
        edge_exp<<<edgeG, blk, 0, stream>>>(ei, P, Mx, Dn, E);
        edge_scatter<<<edgeG, blk, 0, stream>>>(ei, P, Dn, QKVS, Ebf, Agg, E);
        node_update<<<nodeG, blk, 0, stream>>>(Agg, QKVS, Wb + l * 384,
                                               lng + l * 128, lnb + l * 128,
                                               (l == 0) ? nullptr : H1,
                                               (l == 0) ? H1 : Hc, N);
    }
    gemm_bf16_wmma<128, false><<<gemmN, blk, 0, stream>>>(Hc, N, wt_o, bo, OUTb, nullptr);
    final_gate<<<nodeG, blk, 0, stream>>>(x, OUTb, Wg, bg, out, N);
}
